// SurpriseKimiDeltaAttention_50242527428769
// MI455X (gfx1250) — matrix operand order repaired and checked by reading, not compiled, not run on hardware
//
#include <hip/hip_runtime.h>
#include <cstdint>

// ---------------- problem constants ----------------
#define BSZ   2
#define TLEN  4096
#define HD    2048
#define NHD   16
#define DK    128
#define DV    128
#define MROWS (BSZ*TLEN)   // 8192
#define CONVK 4

// ---------------- GEMM tiling ----------------
#define TILE_M 128
#define TILE_N 64
#define TILE_K 64          // two 16x16x32 WMMA K-steps per LDS stage
#define LDSW   72          // padded row stride in ushorts (144B: 16B-aligned, bank-spread)

typedef __attribute__((ext_vector_type(16))) __bf16 v16bf;
typedef __attribute__((ext_vector_type(8)))  float  v8f;
typedef __attribute__((ext_vector_type(4)))  int    v4i;

union FragBF { unsigned int u[8]; v16bf v; };

// gfx1250 async global->LDS path (ASYNCcnt-tracked), guarded so we always compile
#if defined(__has_builtin)
# if __has_builtin(__builtin_amdgcn_global_load_async_to_lds_b128)
#  define HAVE_ASYNC_LDS 1
# endif
# if __has_builtin(__builtin_amdgcn_s_wait_asynccnt)
#  define HAVE_WAIT_ASYNC 1
# endif
#endif

#ifdef HAVE_ASYNC_LDS
#define AS1G __attribute__((address_space(1)))
#define AS3L __attribute__((address_space(3)))
#endif

__device__ __forceinline__ void async_wait_all() {
#ifdef HAVE_ASYNC_LDS
# ifdef HAVE_WAIT_ASYNC
  __builtin_amdgcn_s_wait_asynccnt(0);
# else
  asm volatile("s_wait_asynccnt 0" ::: "memory");
# endif
#endif
}

__device__ __forceinline__ unsigned short f32_to_bf16_rne(float f) {
  unsigned int u = __float_as_uint(f);
  u += 0x7FFFu + ((u >> 16) & 1u);
  return (unsigned short)(u >> 16);
}

__device__ __forceinline__ float wave_sum(float x) {
#pragma unroll
  for (int o = 16; o > 0; o >>= 1) x += __shfl_xor(x, o, 32);
  return x;
}

// ---------------- f32 -> bf16 cast ----------------
__global__ __launch_bounds__(256) void cast_f32_bf16(const float* __restrict__ s,
                                                     unsigned short* __restrict__ d,
                                                     size_t n) {
  size_t i = (size_t)blockIdx.x * blockDim.x + threadIdx.x;
  if (i < n) d[i] = f32_to_bf16_rne(s[i]);
}

// ---------------- bf16 WMMA GEMM: C[M,N] = A[M,K] * W[N,K]^T ----------------
__device__ __forceinline__ void store_acc8(float* __restrict__ C, int ldc,
                                           int row, int col, v8f a) {
  float* p = C + (size_t)row * ldc + col;
#pragma unroll
  for (int i = 0; i < 8; ++i) p[(size_t)i * ldc] = a[i];
}

__global__ __launch_bounds__(256) void wmma_gemm_bf16(
    const unsigned short* __restrict__ Ab, const unsigned short* __restrict__ Wb,
    float* __restrict__ C, int Mtot, int Ntot, int Ktot) {
  (void)Mtot;
  // double-buffered LDS tiles
  __shared__ __align__(16) unsigned short As[2][TILE_M * LDSW];
  __shared__ __align__(16) unsigned short Bs[2][TILE_N * LDSW];

  const int tid  = threadIdx.x;
  const int lane = tid & 31;
  const int wv   = tid >> 5;
  const int wm   = wv & 3;     // wave row (0..3)  -> 32 rows each
  const int wn   = wv >> 2;    // wave col (0..1)  -> 32 cols each
  const int blockM = blockIdx.y * TILE_M;
  const int blockN = blockIdx.x * TILE_N;

  v8f acc00 = {0.f,0.f,0.f,0.f,0.f,0.f,0.f,0.f};
  v8f acc01 = acc00, acc10 = acc00, acc11 = acc00;

  // staging assignment: A = 128 rows x 64 ushorts; 2 thr/row, 32 ushorts (64B) each
  //                     B =  64 rows x 64 ushorts; 4 thr/row, 16 ushorts (32B) each
  const int ra = tid >> 1, ha = tid & 1;
  const int rb = tid >> 2, qb = tid & 3;
  const unsigned short* gA = Ab + (size_t)(blockM + ra) * Ktot + ha * 32;
  const unsigned short* gB = Wb + (size_t)(blockN + rb) * Ktot + qb * 16;
  unsigned short* lA0 = &As[0][ra * LDSW + ha * 32];
  unsigned short* lA1 = &As[1][ra * LDSW + ha * 32];
  unsigned short* lB0 = &Bs[0][rb * LDSW + qb * 16];
  unsigned short* lB1 = &Bs[1][rb * LDSW + qb * 16];

  auto stage = [&](int buf, int k0) {
    unsigned short* dA = buf ? lA1 : lA0;
    unsigned short* dB = buf ? lB1 : lB0;
#ifdef HAVE_ASYNC_LDS
#pragma unroll
    for (int j = 0; j < 4; ++j)
      __builtin_amdgcn_global_load_async_to_lds_b128(
          (AS1G v4i*)(gA + k0 + j * 8), (AS3L v4i*)(dA + j * 8), 0, 0);
#pragma unroll
    for (int j = 0; j < 2; ++j)
      __builtin_amdgcn_global_load_async_to_lds_b128(
          (AS1G v4i*)(gB + k0 + j * 8), (AS3L v4i*)(dB + j * 8), 0, 0);
#else
    const uint4* sA = (const uint4*)(gA + k0);
    uint4 a0v = sA[0], a1v = sA[1], a2v = sA[2], a3v = sA[3];
    *(uint4*)(dA + 0)  = a0v;
    *(uint4*)(dA + 8)  = a1v;
    *(uint4*)(dA + 16) = a2v;
    *(uint4*)(dA + 24) = a3v;
    const uint4* sB = (const uint4*)(gB + k0);
    uint4 b0v = sB[0], b1v = sB[1];
    *(uint4*)(dB + 0) = b0v;
    *(uint4*)(dB + 8) = b1v;
#endif
  };

  stage(0, 0);
  async_wait_all();
  __syncthreads();

  const int hf = lane >> 4;   // lane half selects K sub-range
  const int r0 = lane & 15;   // A: row index / B: column index
  int p = 0;
  for (int k0 = 0; k0 < Ktot; k0 += TILE_K) {
    if (k0 + TILE_K < Ktot) stage(p ^ 1, k0 + TILE_K);

    const unsigned short* Ap = &As[p][0];
    const unsigned short* Bp = &Bs[p][0];
#pragma unroll
    for (int s = 0; s < 2; ++s) {     // two 32-K WMMA steps per stage
      const int koff = s * 32;
      FragBF a0, a1, b0, b1;
#pragma unroll
      for (int i = 0; i < 8; ++i) {   // A 16x32 bf16 per-lane layout (ISA 7.12.2)
        const int kk = koff + ((i < 4) ? 0 : 16) + hf * 8 + 2 * (i & 3);
        a0.u[i] = *(const unsigned int*)&Ap[(wm * 32 + r0)      * LDSW + kk];
        a1.u[i] = *(const unsigned int*)&Ap[(wm * 32 + 16 + r0) * LDSW + kk];
      }
#pragma unroll
      for (int i = 0; i < 8; ++i) {   // B 32x16 bf16 per-lane layout
        const int kk = koff + ((i < 4) ? 0 : 16) + hf * 8 + 2 * (i & 3);
        b0.u[i] = *(const unsigned int*)&Bp[(wn * 32 + r0)      * LDSW + kk];
        b1.u[i] = *(const unsigned int*)&Bp[(wn * 32 + 16 + r0) * LDSW + kk];
      }
      acc00 = __builtin_amdgcn_wmma_f32_16x16x32_bf16(false, a0.v, false, b0.v, (short)0, acc00, false, false);
      acc01 = __builtin_amdgcn_wmma_f32_16x16x32_bf16(false, a0.v, false, b1.v, (short)0, acc01, false, false);
      acc10 = __builtin_amdgcn_wmma_f32_16x16x32_bf16(false, a1.v, false, b0.v, (short)0, acc10, false, false);
      acc11 = __builtin_amdgcn_wmma_f32_16x16x32_bf16(false, a1.v, false, b1.v, (short)0, acc11, false, false);
    }
    async_wait_all();
    __syncthreads();
    p ^= 1;
  }

  store_acc8(C, Ntot, blockM + wm * 32 + 0  + hf * 8, blockN + wn * 32 + 0  + r0, acc00);
  store_acc8(C, Ntot, blockM + wm * 32 + 0  + hf * 8, blockN + wn * 32 + 16 + r0, acc01);
  store_acc8(C, Ntot, blockM + wm * 32 + 16 + hf * 8, blockN + wn * 32 + 0  + r0, acc10);
  store_acc8(C, Ntot, blockM + wm * 32 + 16 + hf * 8, blockN + wn * 32 + 16 + r0, acc11);
}

// ---------------- causal depthwise conv(4) + SiLU ----------------
__global__ __launch_bounds__(256) void conv_silu_kernel(const float* __restrict__ xin,
                                                        const float* __restrict__ wconv,
                                                        float* __restrict__ yout) {
  size_t idx = (size_t)blockIdx.x * blockDim.x + threadIdx.x;
  if (idx >= (size_t)MROWS * HD) return;
  const int c  = (int)(idx % HD);
  const int bt = (int)(idx / HD);
  const int t  = bt % TLEN;
  float acc = 0.f;
#pragma unroll
  for (int i = 0; i < CONVK; ++i) {
    const int ts = t - (CONVK - 1) + i;
    if (ts >= 0) acc += xin[(size_t)(bt - (CONVK - 1) + i) * HD + c] * wconv[c * CONVK + i];
  }
  yout[idx] = acc / (1.f + __expf(-acc));
}

// ---------------- surprise stats, tiny MLPs, gate g, q/k l2-norm ----------------
// one wave per (b,t,h); 32 lanes x float4 covers 128 dims; MLP hidden == 32 lanes
__global__ __launch_bounds__(256) void stats_kernel(
    const float* __restrict__ vhat, const float* __restrict__ v,
    float* __restrict__ qbuf, float* __restrict__ kbuf,
    float* __restrict__ gbuf, float* __restrict__ betabuf,
    const float* __restrict__ Wb1, const float* __restrict__ bb1,
    const float* __restrict__ Wb2, const float* __restrict__ bb2,
    const float* __restrict__ Wa1, const float* __restrict__ ba1,
    const float* __restrict__ Wa2, const float* __restrict__ ba2,
    const float* __restrict__ A_log, const float* __restrict__ dt_bias) {
  const int wid  = (blockIdx.x << 3) + (threadIdx.x >> 5);
  const int lane = threadIdx.x & 31;
  const int h    = wid & (NHD - 1);
  const size_t base = (size_t)wid * DK;
  const int d0 = lane * 4;

  float4 vh = *(const float4*)(vhat + base + d0);
  float4 vv = *(const float4*)(v    + base + d0);
  float4 kk = *(const float4*)(kbuf + base + d0);
  float4 qq = *(const float4*)(qbuf + base + d0);

  const float e0 = vh.x - vv.x, e1 = vh.y - vv.y, e2 = vh.z - vv.z, e3 = vh.w - vv.w;
  float se2 = e0*e0 + e1*e1 + e2*e2 + e3*e3;
  float sl1 = fabsf(e0) + fabsf(e1) + fabsf(e2) + fabsf(e3);
  float shh = vh.x*vh.x + vh.y*vh.y + vh.z*vh.z + vh.w*vh.w;
  float svv = vv.x*vv.x + vv.y*vv.y + vv.z*vv.z + vv.w*vv.w;
  float shv = vh.x*vv.x + vh.y*vv.y + vh.z*vv.z + vh.w*vv.w;
  float sk2 = kk.x*kk.x + kk.y*kk.y + kk.z*kk.z + kk.w*kk.w;
  float sq2 = qq.x*qq.x + qq.y*qq.y + qq.z*qq.z + qq.w*qq.w;
  se2 = wave_sum(se2); sl1 = wave_sum(sl1); shh = wave_sum(shh);
  svv = wave_sum(svv); shv = wave_sum(shv); sk2 = wave_sum(sk2); sq2 = wave_sum(sq2);

  const float s0 = sqrtf(se2 + 1e-6f);
  const float s1 = sl1;
  const float s2 = 1.f - shv / (sqrtf(shh + 1e-6f) * sqrtf(svv + 1e-6f) + 1e-6f);

  float zb = Wb1[lane*3+0]*s0 + Wb1[lane*3+1]*s1 + Wb1[lane*3+2]*s2 + bb1[lane];
  float hb = zb / (1.f + __expf(-zb));
  const float betaLogit = wave_sum(hb * Wb2[lane]) + bb2[0];
  const float beta = 1.f / (1.f + __expf(-betaLogit));

  float za = Wa1[lane*3+0]*s0 + Wa1[lane*3+1]*s1 + Wa1[lane*3+2]*s2 + ba1[lane];
  float ha = za / (1.f + __expf(-za));
  const float amp = wave_sum(ha * Wa2[lane]) + ba2[0];

  if (lane == 0) betabuf[wid] = beta;

  const float rinv = amp / fmaxf(sqrtf(sk2), 1e-12f);   // r*amp = k*rinv
  const float negA = -__expf(A_log[h]);
  const float rsk  = rsqrtf(sk2 + 1e-6f);
  const float rsq  = rsqrtf(sq2 + 1e-6f);

  const float kd[4] = {kk.x, kk.y, kk.z, kk.w};
  const float qd[4] = {qq.x, qq.y, qq.z, qq.w};
  float go[4], qo[4], ko[4];
#pragma unroll
  for (int j = 0; j < 4; ++j) {
    const float xx = kd[j] * rinv + dt_bias[h * DK + d0 + j];
    const float sp = (xx > 20.f) ? xx : log1pf(__expf(xx));
    go[j] = negA * sp;
    qo[j] = qd[j] * rsq;
    ko[j] = kd[j] * rsk;
  }
  *(float4*)(gbuf + base + d0) = make_float4(go[0], go[1], go[2], go[3]);
  *(float4*)(qbuf + base + d0) = make_float4(qo[0], qo[1], qo[2], qo[3]);
  *(float4*)(kbuf + base + d0) = make_float4(ko[0], ko[1], ko[2], ko[3]);
}

// ---------------- sequential delta-rule scan ----------------
// one block per (b,h); thread tid owns column dv=tid of S[DK][DV] (128 VGPRs of state)
// double-buffered LDS broadcast of k / q*scale / exp(g): one barrier per time step
__global__ __launch_bounds__(128) void scan_kernel(
    const float* __restrict__ qn, const float* __restrict__ kn,
    const float* __restrict__ v,  const float* __restrict__ g,
    const float* __restrict__ beta, float* __restrict__ o) {
  const int bh = blockIdx.x;
  const int b = bh / NHD, h = bh % NHD;
  const int tid = threadIdx.x;
  __shared__ __align__(16) float lk[2][DK];
  __shared__ __align__(16) float lq[2][DK];
  __shared__ __align__(16) float le[2][DK];
  float S[DK];
#pragma unroll
  for (int i = 0; i < DK; ++i) S[i] = 0.f;
  const float scale = 0.08838834764831845f;   // DK^-0.5
  const size_t base0   = ((size_t)(b * TLEN) * NHD + h) * DK;
  const size_t strideT = (size_t)NHD * DK;
  const size_t bbase0  = (size_t)(b * TLEN) * NHD + h;

  // prologue: stage t=0 broadcast values
  lk[0][tid] = kn[base0 + tid];
  lq[0][tid] = qn[base0 + tid] * scale;
  le[0][tid] = __expf(g[base0 + tid]);
  __syncthreads();

  int p = 0;
  for (int t = 0; t < TLEN; ++t) {
    const size_t base = base0 + (size_t)t * strideT;
    if (t + 1 < TLEN) {               // prefetch t+1 into the alternate buffer
      const size_t basen = base + strideT;
      lk[p ^ 1][tid] = kn[basen + tid];
      lq[p ^ 1][tid] = qn[basen + tid] * scale;
      le[p ^ 1][tid] = __expf(g[basen + tid]);
    }
    const float vt = v[base + tid];
    const float bt = beta[bbase0 + (size_t)t * NHD];

    float kv = 0.f;
#pragma unroll
    for (int i4 = 0; i4 < DK / 4; ++i4) {
      const float4 k4 = *(const float4*)&lk[p][i4 * 4];
      const float4 e4 = *(const float4*)&le[p][i4 * 4];
      S[i4*4+0] *= e4.x; kv += k4.x * S[i4*4+0];
      S[i4*4+1] *= e4.y; kv += k4.y * S[i4*4+1];
      S[i4*4+2] *= e4.z; kv += k4.z * S[i4*4+2];
      S[i4*4+3] *= e4.w; kv += k4.w * S[i4*4+3];
    }
    const float delta = (vt - kv) * bt;
    float acc = 0.f;
#pragma unroll
    for (int i4 = 0; i4 < DK / 4; ++i4) {
      const float4 k4 = *(const float4*)&lk[p][i4 * 4];
      const float4 q4 = *(const float4*)&lq[p][i4 * 4];
      S[i4*4+0] += k4.x * delta; acc += q4.x * S[i4*4+0];
      S[i4*4+1] += k4.y * delta; acc += q4.y * S[i4*4+1];
      S[i4*4+2] += k4.z * delta; acc += q4.z * S[i4*4+2];
      S[i4*4+3] += k4.w * delta; acc += q4.w * S[i4*4+3];
    }
    o[base + tid] = acc;
    __syncthreads();
    p ^= 1;
  }
}

// ---------------- gated RMSNorm + sigmoid(gate) -> bf16 ----------------
__global__ __launch_bounds__(256) void normgate_kernel(
    const float* __restrict__ o, const float* __restrict__ gate,
    const float* __restrict__ bg2, const float* __restrict__ norm_w,
    unsigned short* __restrict__ og) {
  const int wid  = (blockIdx.x << 3) + (threadIdx.x >> 5);
  const int lane = threadIdx.x & 31;
  const int h    = wid & (NHD - 1);
  const size_t base = (size_t)wid * DV;
  const int d0 = lane * 4;

  const float4 ov = *(const float4*)(o + base + d0);
  float ss = ov.x*ov.x + ov.y*ov.y + ov.z*ov.z + ov.w*ov.w;
  ss = wave_sum(ss);
  const float rn = rsqrtf(ss * (1.f / DV) + 1e-5f);
  const float4 gv = *(const float4*)(gate + base + d0);
  const float od[4] = {ov.x, ov.y, ov.z, ov.w};
  const float gd[4] = {gv.x, gv.y, gv.z, gv.w};
#pragma unroll
  for (int j = 0; j < 4; ++j) {
    const float on = od[j] * rn * norm_w[d0 + j];
    const float gl = gd[j] + bg2[h * DV + d0 + j];
    const float res = on / (1.f + __expf(-gl));
    og[base + d0 + j] = f32_to_bf16_rne(res);
  }
}

// ---------------- host orchestration ----------------
extern "C" void kernel_launch(void* const* d_in, const int* in_sizes, int n_in,
                              void* d_out, int out_size, void* d_ws, size_t ws_size,
                              hipStream_t stream) {
  const float* x      = (const float*)d_in[0];
  const float* Wq     = (const float*)d_in[1];
  const float* Wk     = (const float*)d_in[2];
  const float* Wv     = (const float*)d_in[3];
  const float* conv_q = (const float*)d_in[4];
  const float* conv_k = (const float*)d_in[5];
  const float* conv_v = (const float*)d_in[6];
  const float* A_log  = (const float*)d_in[7];
  const float* dt_bias= (const float*)d_in[8];
  const float* Wsv    = (const float*)d_in[9];
  const float* Wb1    = (const float*)d_in[10];
  const float* bb1    = (const float*)d_in[11];
  const float* Wb2    = (const float*)d_in[12];
  const float* bb2    = (const float*)d_in[13];
  const float* Wa1    = (const float*)d_in[14];
  const float* ba1    = (const float*)d_in[15];
  const float* Wa2    = (const float*)d_in[16];
  const float* ba2    = (const float*)d_in[17];
  const float* Wg1    = (const float*)d_in[18];
  const float* Wg2    = (const float*)d_in[19];
  const float* bg2    = (const float*)d_in[20];
  const float* norm_w = (const float*)d_in[21];
  const float* Wo     = (const float*)d_in[22];
  float* out = (float*)d_out;
  (void)in_sizes; (void)n_in; (void)out_size; (void)ws_size;

  char* ws = (char*)d_ws;
  size_t off = 0;
  auto take = [&](size_t bytes) -> char* {
    char* p = ws + off;
    off += (bytes + 255) & ~(size_t)255;
    return p;
  };
  unsigned short* XB    = (unsigned short*)take((size_t)MROWS * HD * 2);
  unsigned short* WQB   = (unsigned short*)take((size_t)HD * HD * 2);
  unsigned short* WKB   = (unsigned short*)take((size_t)HD * HD * 2);
  unsigned short* WVB   = (unsigned short*)take((size_t)HD * HD * 2);
  unsigned short* WOB   = (unsigned short*)take((size_t)HD * HD * 2);
  unsigned short* WG1B  = (unsigned short*)take((size_t)DV * HD * 2);
  unsigned short* WG2B  = (unsigned short*)take((size_t)HD * DV * 2);
  unsigned short* WSVB  = (unsigned short*)take((size_t)DV * DK * 2);
  float* QRAW = (float*)take((size_t)MROWS * HD * 4);
  float* KRAW = (float*)take((size_t)MROWS * HD * 4);
  float* VRAW = (float*)take((size_t)MROWS * HD * 4);
  float* Qb   = (float*)take((size_t)MROWS * HD * 4);
  float* Kb   = (float*)take((size_t)MROWS * HD * 4);
  float* Vb   = (float*)take((size_t)MROWS * HD * 4);
  unsigned short* BF16S = (unsigned short*)take((size_t)MROWS * HD * 2);
  float* BETA = (float*)take((size_t)MROWS * NHD * 4);
  float* GMID = (float*)take((size_t)MROWS * DV * 4);
  unsigned short* GMIDB = (unsigned short*)take((size_t)MROWS * DV * 2);
  // reuse: projections are dead after the conv / stats stages
  float* VHAT = QRAW;   // [M*NH, DV] f32 (consumed by stats before scan)
  float* Gg   = KRAW;   // decay gate g
  float* GATE = VRAW;   // sigmoid-gate pre-activation
  float* Obuf = QRAW;   // scan output (VHAT already consumed)

  auto cast = [&](const float* s, unsigned short* d, size_t n) {
    cast_f32_bf16<<<dim3((unsigned)((n + 255) / 256)), dim3(256), 0, stream>>>(s, d, n);
  };
  auto gemm = [&](const unsigned short* Ap, const unsigned short* Wp, float* Cp,
                  int Mt, int Nt, int Kt) {
    wmma_gemm_bf16<<<dim3(Nt / TILE_N, Mt / TILE_M), dim3(256), 0, stream>>>(Ap, Wp, Cp, Mt, Nt, Kt);
  };

  // bf16 staging
  cast(x,   XB,   (size_t)MROWS * HD);
  cast(Wq,  WQB,  (size_t)HD * HD);
  cast(Wk,  WKB,  (size_t)HD * HD);
  cast(Wv,  WVB,  (size_t)HD * HD);
  cast(Wo,  WOB,  (size_t)HD * HD);
  cast(Wg1, WG1B, (size_t)DV * HD);
  cast(Wg2, WG2B, (size_t)HD * DV);
  cast(Wsv, WSVB, (size_t)DV * DK);

  // q/k/v projections (WMMA bf16, fp32 accumulate)
  gemm(XB, WQB, QRAW, MROWS, HD, HD);
  gemm(XB, WKB, KRAW, MROWS, HD, HD);
  gemm(XB, WVB, VRAW, MROWS, HD, HD);

  // causal conv + SiLU (fp32)
  const unsigned convBlocks = (unsigned)(((size_t)MROWS * HD + 255) / 256);
  conv_silu_kernel<<<convBlocks, 256, 0, stream>>>(QRAW, conv_q, Qb);
  conv_silu_kernel<<<convBlocks, 256, 0, stream>>>(KRAW, conv_k, Kb);
  conv_silu_kernel<<<convBlocks, 256, 0, stream>>>(VRAW, conv_v, Vb);

  // v_hat = k @ Wsv^T as a [M*NH, DV, DK] WMMA GEMM
  cast(Kb, BF16S, (size_t)MROWS * HD);
  gemm(BF16S, WSVB, VHAT, MROWS * NHD, DV, DK);

  // surprise stats + MLPs + gates + q/k normalization (in-place on Qb/Kb)
  stats_kernel<<<dim3(MROWS * NHD / 8), dim3(256), 0, stream>>>(
      VHAT, Vb, Qb, Kb, Gg, BETA, Wb1, bb1, Wb2, bb2, Wa1, ba1, Wa2, ba2, A_log, dt_bias);

  // gate path: x@Wg1^T -> bf16 -> @Wg2^T
  gemm(XB, WG1B, GMID, MROWS, DV, HD);
  cast(GMID, GMIDB, (size_t)MROWS * DV);
  gemm(GMIDB, WG2B, GATE, MROWS, HD, DV);

  // sequential delta-rule scan (fp32 state in VGPRs)
  scan_kernel<<<dim3(BSZ * NHD), dim3(DK), 0, stream>>>(Qb, Kb, Vb, Gg, BETA, Obuf);

  // gated RMSNorm -> bf16, then output projection (WMMA) straight into d_out
  normgate_kernel<<<dim3(MROWS * NHD / 8), dim3(256), 0, stream>>>(Obuf, GATE, bg2, norm_w, BF16S);
  gemm(BF16S, WOB, out, MROWS, HD, HD);
}
